// DeepSeekV3Model_28037546508965
// MI455X (gfx1250) — compile-verified, matrix operand order checked by reading, not run on hardware
//
#include <hip/hip_runtime.h>
#include <cstdint>

// ---- problem constants (match reference) ----
#define B_     2
#define S_     2048
#define DIM_   2048
#define NH_    16
#define QLORA  1536
#define KVLORA 512
#define NOPE_  128
#define ROPE_  64
#define VH_    128
#define QKHD   192          // NOPE + ROPE
#define EPS_   1e-6f

typedef __attribute__((ext_vector_type(16))) __bf16  v16bf;
typedef __attribute__((ext_vector_type(8)))  float   v8f;
typedef __attribute__((ext_vector_type(4)))  unsigned int u32x4;

__device__ __forceinline__ uint16_t f2bf(float f) {
  uint32_t u = __builtin_bit_cast(uint32_t, f);
  uint32_t r = (u + 0x7FFFu + ((u >> 16) & 1u)) >> 16;   // RNE
  return (uint16_t)r;
}

union Frag { v16bf v; u32x4 q[2]; };

// lane fragment: 8 contiguous bf16 at lo, 8 contiguous bf16 at hi (each 16B aligned)
__device__ __forceinline__ v16bf ld_frag(const uint16_t* lo, const uint16_t* hi) {
  Frag f;
  f.q[0] = *reinterpret_cast<const u32x4*>(lo);
  f.q[1] = *reinterpret_cast<const u32x4*>(hi);
  return f.v;
}

__device__ __forceinline__ v8f wmma_bf16(v16bf a, v16bf b, v8f c) {
  return __builtin_amdgcn_wmma_f32_16x16x32_bf16(false, a, false, b, (short)0, c, false, false);
}

// CDNA5 async copy: 16B per lane, global -> LDS, tracked by ASYNCcnt
__device__ __forceinline__ void async_b128_to_lds(uint32_t lds_off, const void* gptr) {
  asm volatile("global_load_async_to_lds_b128 %0, %1, off"
               :: "v"(lds_off), "v"((uint64_t)(uintptr_t)gptr)
               : "memory");
}

// ---------------- elementwise / conversion kernels ----------------

__global__ void k_f32_to_bf16(const float* __restrict__ in, uint16_t* __restrict__ out, long n) {
  long i = (long)blockIdx.x * 256 + threadIdx.x;
  if (i < n) out[i] = f2bf(in[i]);
}

// out[n*K + k] = in[k*N + n]   (store W^T as bf16 so WMMA B-fragments load contiguously)
__global__ void k_transpose_bf16(const float* __restrict__ in, uint16_t* __restrict__ out,
                                 int K, int N) {
  long i = (long)blockIdx.x * 256 + threadIdx.x;
  if (i >= (long)K * N) return;
  int k = (int)(i / N), n = (int)(i % N);
  out[(long)n * K + k] = f2bf(in[i]);
}

// one block per row: rmsnorm + weight + f32->bf16
__global__ void k_rmsnorm_bf16(const float* __restrict__ in, int in_stride,
                               const float* __restrict__ w, int D,
                               uint16_t* __restrict__ out, int out_stride) {
  __shared__ float red[256];
  const long row = blockIdx.x;
  const float* x = in + row * in_stride;
  float s = 0.f;
  for (int d = threadIdx.x; d < D; d += 256) { float v = x[d]; s += v * v; }
  red[threadIdx.x] = s; __syncthreads();
  for (int off = 128; off; off >>= 1) {
    if (threadIdx.x < (unsigned)off) red[threadIdx.x] += red[threadIdx.x + off];
    __syncthreads();
  }
  float r = rsqrtf(red[0] / (float)D + EPS_);
  uint16_t* o = out + (long)row * out_stride;
  for (int d = threadIdx.x; d < D; d += 256) o[d] = f2bf(x[d] * r * w[d]);
}

// q f32 [B*S, NH*192] -> Q bf16 [B, NH, S, 192], RoPE on d in [128,192), * SCALE
__global__ void k_build_q(const float* __restrict__ q, const float* __restrict__ cosT,
                          const float* __restrict__ sinT, uint16_t* __restrict__ Q,
                          float scale) {
  long idx = (long)blockIdx.x * 256 + threadIdx.x;
  if (idx >= (long)B_ * S_ * NH_ * QKHD) return;
  int d = (int)(idx % QKHD); long t = idx / QKHD;
  int h = (int)(t % NH_); t /= NH_;
  int s = (int)(t % S_);  int b = (int)(t / S_);
  const float* row = q + (((long)(b * S_ + s)) * NH_ + h) * QKHD;
  float v;
  if (d < NOPE_) v = row[d];
  else {
    int j = (d - NOPE_) >> 1;
    float xr = row[NOPE_ + 2 * j], xi = row[NOPE_ + 2 * j + 1];
    float c = cosT[s * (ROPE_ / 2) + j], sn = sinT[s * (ROPE_ / 2) + j];
    v = ((d - NOPE_) & 1) ? (xr * sn + xi * c) : (xr * c - xi * sn);
  }
  Q[(((long)(b * NH_ + h)) * S_ + s) * QKHD + d] = f2bf(v * scale);
}

// RoPE on k_pe slice of kv f32 [B*S, 576] -> kpe f32 [B*S, 64]
__global__ void k_rope_kpe(const float* __restrict__ kv, const float* __restrict__ cosT,
                           const float* __restrict__ sinT, float* __restrict__ kpe) {
  long idx = (long)blockIdx.x * 256 + threadIdx.x;
  if (idx >= (long)B_ * S_ * (ROPE_ / 2)) return;
  int j = (int)(idx % (ROPE_ / 2)); long t = idx / (ROPE_ / 2);
  int s = (int)(t % S_); int b = (int)(t / S_);
  const float* row = kv + (long)(b * S_ + s) * (KVLORA + ROPE_) + KVLORA;
  float xr = row[2 * j], xi = row[2 * j + 1];
  float c = cosT[s * (ROPE_ / 2) + j], sn = sinT[s * (ROPE_ / 2) + j];
  float* o = kpe + (long)(b * S_ + s) * ROPE_;
  o[2 * j]     = xr * c - xi * sn;
  o[2 * j + 1] = xr * sn + xi * c;
}

// kvb f32 [B*S, NH*256] + kpe -> K bf16 [B,NH,S,192], V^T bf16 [B,NH,128,S]
__global__ void k_build_kv(const float* __restrict__ kvb, const float* __restrict__ kpe,
                           uint16_t* __restrict__ Kt, uint16_t* __restrict__ Vt) {
  long idx = (long)blockIdx.x * 256 + threadIdx.x;
  if (idx >= (long)B_ * NH_ * S_ * 320) return;
  int c = (int)(idx % 320); long t = idx / 320;
  int s = (int)(t % S_); t /= S_;
  int h = (int)(t % NH_); int b = (int)(t / NH_);
  long bh = (long)b * NH_ + h;
  long rowoff = ((long)(b * S_ + s)) * (NH_ * 256) + h * 256;
  if (c < NOPE_)
    Kt[(bh * S_ + s) * QKHD + c] = f2bf(kvb[rowoff + c]);
  else if (c < QKHD)
    Kt[(bh * S_ + s) * QKHD + c] = f2bf(kpe[(long)(b * S_ + s) * ROPE_ + (c - NOPE_)]);
  else {
    int dv = c - QKHD;
    Vt[(bh * VH_ + dv) * S_ + s] = f2bf(kvb[rowoff + NOPE_ + dv]);
  }
}

// ---------------- bf16 WMMA GEMM: C[M,N] f32 = A[M,K] x Bt[N,K]^T ----------------
// block = 256 threads (8 waves); wave w: rows [by*128 + w*16, +16), cols [bx*64, +64)
// B tile (64 cols x 32 k) staged once per block into LDS via CDNA5 async copy
// (double-buffered, ASYNCcnt). A fragments stream from global (unique per wave).
__global__ void __launch_bounds__(256)
k_gemm_bf16(const uint16_t* __restrict__ A, const uint16_t* __restrict__ Bt,
            float* __restrict__ C, int M, int N, int K) {
  __shared__ uint16_t bs[2][64][40];   // rows padded to 80B: 16B-aligned, conflict-free frags
  const int tid = threadIdx.x;
  const int wave = tid >> 5;
  const int lane = tid & 31;
  const int m = lane & 15, hi = lane >> 4;
  const int row0 = blockIdx.y * 128 + wave * 16;
  const int col0 = blockIdx.x * 64;
  const int klo = hi * 8, khi = 16 + hi * 8;

  // staging role: thread t loads 16B: B row (col) = t>>2, 16B segment = t&3
  const int srow = tid >> 2;
  const int sseg = tid & 3;
  const uint16_t* bsrc = Bt + (long)(col0 + srow) * K + sseg * 8;
  const uint32_t soff0 = (uint32_t)(uintptr_t)&bs[0][srow][sseg * 8];
  const uint32_t sbufstride = (uint32_t)(64 * 40 * 2);

  const uint16_t* arow = A + (long)(row0 + m) * K;

  v8f acc[4];
  v8f zz = {0.f, 0.f, 0.f, 0.f, 0.f, 0.f, 0.f, 0.f};
  for (int j = 0; j < 4; ++j) acc[j] = zz;

  auto do_step = [&](int i) {
    const int k = i << 5;
    v16bf a = ld_frag(arow + k + klo, arow + k + khi);
    const uint16_t (*bb)[40] = bs[i & 1];
#pragma unroll
    for (int j = 0; j < 4; ++j) {
      v16bf b = ld_frag(&bb[j * 16 + m][klo], &bb[j * 16 + m][khi]);
      acc[j] = wmma_bf16(a, b, acc[j]);
    }
  };

  const int nsteps = K >> 5;
  async_b128_to_lds(soff0, bsrc);                       // stage step 0 -> buf 0
  for (int i = 0; i < nsteps - 1; ++i) {
    async_b128_to_lds(soff0 + ((i + 1) & 1) * sbufstride, bsrc + (long)(i + 1) * 32);
    asm volatile("s_wait_asynccnt 1" ::: "memory");     // step i landed in LDS
    __syncthreads();                                    // visible to all waves
    do_step(i);
    __syncthreads();                                    // done reading before restage
  }
  asm volatile("s_wait_asynccnt 0" ::: "memory");
  __syncthreads();
  do_step(nsteps - 1);

#pragma unroll
  for (int j = 0; j < 4; ++j)
#pragma unroll
    for (int r = 0; r < 8; ++r)
      C[(long)(row0 + r + 8 * hi) * N + col0 + j * 16 + m] = acc[j][r];
}

// ---------------- flash attention (causal, online softmax, WMMA) ----------------
// grid (S/128, NH, B); block 256 (8 waves); wave -> 16 query rows.
// K (32x192) and V (128x32) tiles staged block-cooperatively via async-to-LDS,
// double-buffered; all waves run to the block-max tile count (uniform barriers),
// causally-finished waves skip the compute phase.
__global__ void __launch_bounds__(256)
k_mla_attn(const uint16_t* __restrict__ Q,   // [B,NH,S,192], SCALE folded in
           const uint16_t* __restrict__ Kt,  // [B,NH,S,192]
           const uint16_t* __restrict__ Vt,  // [B,NH,128,S]
           uint16_t* __restrict__ O) {       // [B*S, NH*128] bf16
  __shared__ uint16_t ldsK[2][32][200];      // key rows padded to 400B (16B aligned)
  __shared__ uint16_t ldsV[2][128][40];      // d rows padded to 80B
  __shared__ uint16_t ldsP[8][16][40];       // per-wave P transpose staging
  const int tid = threadIdx.x;
  const int wave = tid >> 5;
  const int lane = tid & 31;
  const int m = lane & 15, hi = lane >> 4;
  const int h = blockIdx.y, b = blockIdx.z;
  const long bh = (long)b * NH_ + h;
  const int Q0 = blockIdx.x * 128;
  const int q0 = Q0 + wave * 16;
  const int klo = hi * 8, khi = 16 + hi * 8;

  // ---- per-thread staging descriptors: 3 K segs + 2 V segs per tile ----
  uint32_t kldsoff[3]; const uint16_t* kgsrc[3];
#pragma unroll
  for (int j = 0; j < 3; ++j) {
    int s = tid + j * 256;                  // 768 16B segments: 32 rows x 24 segs
    int row = s / 24, seg = s - row * 24;
    kldsoff[j] = (uint32_t)(uintptr_t)&ldsK[0][row][seg * 8];
    kgsrc[j] = Kt + (bh * S_ + row) * (long)QKHD + seg * 8;
  }
  uint32_t vldsoff[2]; const uint16_t* vgsrc[2];
#pragma unroll
  for (int j = 0; j < 2; ++j) {
    int s = tid + j * 256;                  // 512 16B segments: 128 rows x 4 segs
    int row = s >> 2, seg = s & 3;
    vldsoff[j] = (uint32_t)(uintptr_t)&ldsV[0][row][seg * 8];
    vgsrc[j] = Vt + (bh * VH_ + row) * (long)S_ + seg * 8;
  }
  const uint32_t kbufstride = (uint32_t)(32 * 200 * 2);
  const uint32_t vbufstride = (uint32_t)(128 * 40 * 2);

  auto stage = [&](int kt) {               // 5 async instructions per wave
    const int buf = kt & 1;
    const long koff = (long)(kt << 5) * QKHD;
    const int voff = kt << 5;
#pragma unroll
    for (int j = 0; j < 3; ++j)
      async_b128_to_lds(kldsoff[j] + buf * kbufstride, kgsrc[j] + koff);
#pragma unroll
    for (int j = 0; j < 2; ++j)
      async_b128_to_lds(vldsoff[j] + buf * vbufstride, vgsrc[j] + voff);
  };

  // ---- Q fragments (6 chunks of k=32 over d=192), resident in registers ----
  const uint16_t* Qb = Q + (bh * S_ + q0 + m) * (long)QKHD;
  v16bf qf[6];
#pragma unroll
  for (int c = 0; c < 6; ++c) qf[c] = ld_frag(Qb + c * 32 + klo, Qb + c * 32 + khi);

  v8f z = {0.f, 0.f, 0.f, 0.f, 0.f, 0.f, 0.f, 0.f};
  v8f acc[8];
#pragma unroll
  for (int i = 0; i < 8; ++i) acc[i] = z;
  float mrow[8], lrow[8];
#pragma unroll
  for (int r = 0; r < 8; ++r) { mrow[r] = -1e30f; lrow[r] = 0.f; }

  auto compute = [&](int kt) {
    const int kb = kt << 5;
    const int buf = kt & 1;
    // ---- S = Q K^T (two 16-key sub-tiles) from LDS ----
    v8f s0 = z, s1 = z;
#pragma unroll
    for (int c = 0; c < 6; ++c) {
      v16bf b0 = ld_frag(&ldsK[buf][m][c * 32 + klo],      &ldsK[buf][m][c * 32 + khi]);
      v16bf b1 = ld_frag(&ldsK[buf][16 + m][c * 32 + klo], &ldsK[buf][16 + m][c * 32 + khi]);
      s0 = wmma_bf16(qf[c], b0, s0);
      s1 = wmma_bf16(qf[c], b1, s1);
    }
    // ---- causal mask (lane owns key col = kb+sub*16+m, row = q0+r+8*hi)
    if (kb + 31 > q0) {
#pragma unroll
      for (int r = 0; r < 8; ++r) {
        int rowg = q0 + r + 8 * hi;
        if (kb + m > rowg)      s0[r] = -1e30f;
        if (kb + 16 + m > rowg) s1[r] = -1e30f;
      }
    }
    // ---- online softmax (row reductions across the 16-lane half) ----
    float a_r[8];
#pragma unroll
    for (int r = 0; r < 8; ++r) {
      float mx = fmaxf(s0[r], s1[r]);
      for (int off = 1; off < 16; off <<= 1) mx = fmaxf(mx, __shfl_xor(mx, off, 32));
      float mnew = fmaxf(mrow[r], mx);
      float alpha = __expf(mrow[r] - mnew);
      float p0 = __expf(s0[r] - mnew);
      float p1 = __expf(s1[r] - mnew);
      s0[r] = p0; s1[r] = p1;
      float rs = p0 + p1;
      for (int off = 1; off < 16; off <<= 1) rs += __shfl_xor(rs, off, 32);
      lrow[r] = lrow[r] * alpha + rs;
      mrow[r] = mnew;
      a_r[r] = alpha;
    }
#pragma unroll
    for (int dt = 0; dt < 8; ++dt)
#pragma unroll
      for (int r = 0; r < 8; ++r) acc[dt][r] = acc[dt][r] * a_r[r];
    // ---- transpose P (C layout -> A layout) through per-wave LDS, as bf16 ----
#pragma unroll
    for (int r = 0; r < 8; ++r) {
      ldsP[wave][r + 8 * hi][m]      = f2bf(s0[r]);
      ldsP[wave][r + 8 * hi][16 + m] = f2bf(s1[r]);
    }
    asm volatile("s_wait_dscnt 0" ::: "memory");  // wave-local LDS RAW; DS in-order per wave
    v16bf pf = ld_frag(&ldsP[wave][m][klo], &ldsP[wave][m][khi]);
    // ---- O += P V from LDS ----
#pragma unroll
    for (int dt = 0; dt < 8; ++dt) {
      v16bf bv = ld_frag(&ldsV[buf][dt * 16 + m][klo], &ldsV[buf][dt * 16 + m][khi]);
      acc[dt] = wmma_bf16(pf, bv, acc[dt]);
    }
  };

  const int ntmax = (Q0 + 128 + 31) >> 5;   // block-max tile count (wave 7's need)
  stage(0);
  for (int kt = 0; kt < ntmax; ++kt) {
    if (kt + 1 < ntmax) {
      stage(kt + 1);
      asm volatile("s_wait_asynccnt 5" ::: "memory");  // tile kt (5 oldest ops) landed
    } else {
      asm volatile("s_wait_asynccnt 0" ::: "memory");
    }
    __syncthreads();                                   // tile kt visible to all waves
    if ((kt << 5) < q0 + 16) compute(kt);              // skip beyond this wave's causal range
    __syncthreads();                                   // all readers done before restage
  }

  // ---- epilogue: O /= l, store bf16 [B*S, NH*128] ----
#pragma unroll
  for (int r = 0; r < 8; ++r) {
    float inv = 1.f / lrow[r];
    long row = (long)b * S_ + q0 + r + 8 * hi;
#pragma unroll
    for (int dt = 0; dt < 8; ++dt)
      O[row * (NH_ * VH_) + h * VH_ + dt * 16 + m] = f2bf(acc[dt][r] * inv);
  }
}

// ---------------- host-side orchestration ----------------

extern "C" void kernel_launch(void* const* d_in, const int* in_sizes, int n_in,
                              void* d_out, int out_size, void* d_ws, size_t ws_size,
                              hipStream_t stream) {
  const float* x        = (const float*)d_in[0];
  const float* cosT     = (const float*)d_in[1];
  const float* sinT     = (const float*)d_in[2];
  const float* wq_a     = (const float*)d_in[3];
  const float* q_norm_w = (const float*)d_in[4];
  const float* wq_b     = (const float*)d_in[5];
  const float* wkv_a    = (const float*)d_in[6];
  const float* kv_norm  = (const float*)d_in[7];
  const float* wkv_b    = (const float*)d_in[8];
  const float* wo       = (const float*)d_in[9];
  float* out = (float*)d_out;

  const long BS = (long)B_ * S_;                 // 4096 rows
  char* wsp = (char*)d_ws;
  auto alloc = [&](size_t bytes) { char* p = wsp; wsp += (bytes + 255) & ~(size_t)255; return p; };

  uint16_t* x_bf   = (uint16_t*)alloc(BS * DIM_ * 2);
  uint16_t* wqa_t  = (uint16_t*)alloc((size_t)DIM_ * QLORA * 2);
  uint16_t* wqb_t  = (uint16_t*)alloc((size_t)QLORA * NH_ * QKHD * 2);
  uint16_t* wkva_t = (uint16_t*)alloc((size_t)DIM_ * (KVLORA + ROPE_) * 2);
  uint16_t* wkvb_t = (uint16_t*)alloc((size_t)KVLORA * NH_ * 256 * 2);
  uint16_t* wo_t   = (uint16_t*)alloc((size_t)NH_ * VH_ * DIM_ * 2);
  float*    qa_f   = (float*)alloc(BS * QLORA * 4);
  uint16_t* qa_n   = (uint16_t*)alloc(BS * QLORA * 2);
  float*    q_f    = (float*)alloc(BS * NH_ * QKHD * 4);
  uint16_t* Q_bf   = (uint16_t*)alloc(BS * NH_ * QKHD * 2);
  float*    kv_f   = (float*)alloc(BS * (KVLORA + ROPE_) * 4);
  uint16_t* kvlat  = (uint16_t*)alloc(BS * KVLORA * 2);
  float*    kpe    = (float*)alloc(BS * ROPE_ * 4);
  float*    kvb_f  = (float*)alloc(BS * NH_ * 256 * 4);
  uint16_t* K_bf   = (uint16_t*)alloc(BS * NH_ * QKHD * 2);
  uint16_t* Vt_bf  = (uint16_t*)alloc(BS * NH_ * VH_ * 2);
  uint16_t* O_bf   = (uint16_t*)alloc(BS * NH_ * VH_ * 2);

  const float SCALE = rsqrtf((float)QKHD);   // host-side: 192^-0.5
  auto cdiv = [](long n, long d) { return (unsigned)((n + d - 1) / d); };

  // x -> bf16
  k_f32_to_bf16<<<cdiv(BS * DIM_, 256), 256, 0, stream>>>(x, x_bf, BS * DIM_);
  // weights -> transposed bf16
  k_transpose_bf16<<<cdiv((long)DIM_ * QLORA, 256), 256, 0, stream>>>(wq_a, wqa_t, DIM_, QLORA);
  k_transpose_bf16<<<cdiv((long)QLORA * NH_ * QKHD, 256), 256, 0, stream>>>(wq_b, wqb_t, QLORA, NH_ * QKHD);
  k_transpose_bf16<<<cdiv((long)DIM_ * (KVLORA + ROPE_), 256), 256, 0, stream>>>(wkv_a, wkva_t, DIM_, KVLORA + ROPE_);
  k_transpose_bf16<<<cdiv((long)KVLORA * NH_ * 256, 256), 256, 0, stream>>>(wkv_b, wkvb_t, KVLORA, NH_ * 256);
  k_transpose_bf16<<<cdiv((long)NH_ * VH_ * DIM_, 256), 256, 0, stream>>>(wo, wo_t, NH_ * VH_, DIM_);

  // q_a = x @ wq_a   [4096 x 1536]
  k_gemm_bf16<<<dim3(QLORA / 64, BS / 128), 256, 0, stream>>>(x_bf, wqa_t, qa_f, (int)BS, QLORA, DIM_);
  // rmsnorm(q_a) -> bf16
  k_rmsnorm_bf16<<<(unsigned)BS, 256, 0, stream>>>(qa_f, QLORA, q_norm_w, QLORA, qa_n, QLORA);
  // q = qa_n @ wq_b   [4096 x 3072]
  k_gemm_bf16<<<dim3((NH_ * QKHD) / 64, BS / 128), 256, 0, stream>>>(qa_n, wqb_t, q_f, (int)BS, NH_ * QKHD, QLORA);
  // RoPE + head-major transpose + scale -> Q bf16
  k_build_q<<<cdiv(BS * NH_ * QKHD, 256), 256, 0, stream>>>(q_f, cosT, sinT, Q_bf, SCALE);

  // kv = x @ wkv_a   [4096 x 576]
  k_gemm_bf16<<<dim3((KVLORA + ROPE_) / 64, BS / 128), 256, 0, stream>>>(x_bf, wkva_t, kv_f, (int)BS, KVLORA + ROPE_, DIM_);
  // rmsnorm(kv_lat) -> bf16 ; RoPE(k_pe)
  k_rmsnorm_bf16<<<(unsigned)BS, 256, 0, stream>>>(kv_f, KVLORA + ROPE_, kv_norm, KVLORA, kvlat, KVLORA);
  k_rope_kpe<<<cdiv(BS * (ROPE_ / 2), 256), 256, 0, stream>>>(kv_f, cosT, sinT, kpe);
  // kvb = kvlat @ wkv_b   [4096 x 4096]
  k_gemm_bf16<<<dim3((NH_ * 256) / 64, BS / 128), 256, 0, stream>>>(kvlat, wkvb_t, kvb_f, (int)BS, NH_ * 256, KVLORA);
  // build K [B,NH,S,192], V^T [B,NH,128,S]
  k_build_kv<<<cdiv((long)B_ * NH_ * S_ * 320, 256), 256, 0, stream>>>(kvb_f, kpe, K_bf, Vt_bf);

  // flash attention
  k_mla_attn<<<dim3(S_ / 128, NH_, B_), 256, 0, stream>>>(Q_bf, K_bf, Vt_bf, O_bf);

  // out = attn @ wo   [4096 x 2048]
  k_gemm_bf16<<<dim3(DIM_ / 64, BS / 128), 256, 0, stream>>>(O_bf, wo_t, out, (int)BS, DIM_, NH_ * VH_);
}
